// GlobalNewsLongEncoder_6279242187082
// MI455X (gfx1250) — matrix operand order repaired and checked by reading, not compiled, not run on hardware
//
#include <hip/hip_runtime.h>
#include <hip/hip_bf16.h>

// ---------------------------------------------------------------------------
// GlobalNewsLongEncoder for MI455X (gfx1250, wave32, WMMA).
// Pipeline: traverse (gather+dot, wave/item) -> QKV GEMM (f16 WMMA, f32 acc)
//           -> per-head attention (WMMA S = Q*K^T, softmax, WMMA O = P*V)
//           -> LN1 + tanh attention pool + LN2 (VALU + LDS reductions).
// ---------------------------------------------------------------------------

typedef __attribute__((ext_vector_type(16))) _Float16 v16h;
typedef __attribute__((ext_vector_type(8)))  _Float16 v8h;
typedef __attribute__((ext_vector_type(8)))  float    v8f;

#define B_     16
#define NN_    50
#define D_     400
#define H_     20
#define DH_    20
#define L_     500     // 50 * SLOTS
#define LP     512     // padded sequence length (32 M-tiles)
#define DP     416     // padded D (13 K-steps of 32)
#define DHP    32      // padded head dim (1 K-step of 32)
#define K_     10
#define STEPS_ 6
#define SLOTS_ 10
#define ATTH   128

// ---------------------------- WMMA helpers ---------------------------------

__device__ __forceinline__ v8f wmma_f16(v16h a, v16h b, v8f c) {
  // v_wmma_f32_16x16x32_f16  (codegen-confirmed builtin, 8 args)
  return __builtin_amdgcn_wmma_f32_16x16x32_f16(
      /*neg_a=*/false, a, /*neg_b=*/false, b,
      /*c_mod=*/(short)0, c, /*reuse_a=*/false, /*reuse_b=*/false);
}

// A-matrix fragment (16x32 f16, row-major source, leading dim ld in halves).
// Per ISA layout: lanes 0-15 hold M=lane, K groups {0..7, 16..23};
//                 lanes 16-31 hold M=lane-16, K groups {8..15, 24..31}.
__device__ __forceinline__ v16h load_frag_A(const _Float16* base, int m0, int k0, int ld) {
  int lane = threadIdx.x & 31;
  const _Float16* p = base + (size_t)(m0 + (lane & 15)) * ld + k0 + ((lane >> 4) << 3);
  v8h lo = *(const v8h*)p;        // K = koff .. koff+7
  v8h hi = *(const v8h*)(p + 16); // K = koff+16 .. koff+23
  v16h r;
#pragma unroll
  for (int i = 0; i < 8; ++i) { r[i] = lo[i]; r[i + 8] = hi[i]; }
  return r;
}

// B-matrix fragment (32x16 f16) loaded from a TRANSPOSED source Bt[n][k]
// (leading dim ld). lanes 0-15: N=lane, K {0..7, 16..23}; lanes 16-31: N=lane-16, K {8..15, 24..31}.
__device__ __forceinline__ v16h load_frag_B(const _Float16* baseT, int k0, int n0, int ld) {
  int lane = threadIdx.x & 31;
  const _Float16* p = baseT + (size_t)(n0 + (lane & 15)) * ld + k0 + ((lane >> 4) << 3);
  v8h lo = *(const v8h*)p;
  v8h hi = *(const v8h*)(p + 16);
  v16h r;
#pragma unroll
  for (int i = 0; i < 8; ++i) { r[i] = lo[i]; r[i + 8] = hi[i]; }
  return r;
}

__device__ __forceinline__ float waveSum(float v) {
#pragma unroll
  for (int m = 16; m >= 1; m >>= 1) v += __shfl_xor(v, m, 32);
  return v;
}

// ------------------------- workspace init ----------------------------------

__global__ void zero_h_kernel(_Float16* p, long long n) {
  long long i  = (long long)blockIdx.x * blockDim.x + threadIdx.x;
  long long st = (long long)gridDim.x * blockDim.x;
  for (; i < n; i += st) p[i] = (_Float16)0.f;
}

// Wt[w][n][k] = W_w[k][n] (transposed, K padded to DP with zeros).
// The 1/sqrt(DH) attention scale is folded into Wq.
__global__ void convw_kernel(const float* __restrict__ Wq, const float* __restrict__ Wk,
                             const float* __restrict__ Wv, _Float16* __restrict__ Wt) {
  long long total = 3LL * D_ * DP;
  long long i  = (long long)blockIdx.x * blockDim.x + threadIdx.x;
  long long st = (long long)gridDim.x * blockDim.x;
  const float qs = 0.22360679774997896f; // 1/sqrt(20)
  for (; i < total; i += st) {
    int k = (int)(i % DP);
    long long r = i / DP;
    int n = (int)(r % D_);
    int w = (int)(r / D_);
    float v = 0.f;
    if (k < D_) {
      const float* W = (w == 0) ? Wq : ((w == 1) ? Wk : Wv);
      v = W[(size_t)k * D_ + n];
      if (w == 0) v *= qs;
    }
    Wt[i] = (_Float16)v;
  }
}

// --------------------------- traverse --------------------------------------
// One wave32 per (b,n) item. 6 sequential hops, K=10 dot products of D=400.
__global__ void traverse_kernel(const float* __restrict__ news,
                                const float* __restrict__ table,
                                const int* __restrict__ click,
                                const int* __restrict__ nbrt,
                                _Float16* __restrict__ Xh) {
  int wid  = (blockIdx.x * blockDim.x + threadIdx.x) >> 5;
  int lane = threadIdx.x & 31;
  if (wid >= B_ * NN_) return;
  int b = wid / NN_, n = wid % NN_;

  const float* nv = news + (size_t)wid * D_;
  float nreg[13];
  int nzl = 0;
#pragma unroll
  for (int i = 0; i < 13; ++i) {
    int d = lane + i * 32;
    float v = (d < D_) ? nv[d] : 0.f;
    nreg[i] = v;
    nzl |= (v != 0.f) ? 1 : 0;
  }
#pragma unroll
  for (int m = 16; m >= 1; m >>= 1) nzl |= __shfl_xor(nzl, m, 32);
  float validf = nzl ? 1.f : 0.f;

  int idx = click[wid];
  bool dead = false;

  for (int step = 0; step < STEPS_; ++step) {
    const float* vg = table + (size_t)(idx - 1) * (K_ * D_);
    float sc[K_];
#pragma unroll
    for (int k = 0; k < K_; ++k) {
      const float* row = vg + k * D_;
      float p = 0.f;
#pragma unroll
      for (int i = 0; i < 13; ++i) {
        int d = lane + i * 32;
        p += (d < D_) ? row[d] * nreg[i] : 0.f;
      }
      sc[k] = waveSum(p); // every lane holds the full dot product
    }
    // uniform per-wave selection logic (first-max semantics like jnp.argmax)
    int mx_i = 0; float mx_v = sc[0];
#pragma unroll
    for (int k = 1; k < K_; ++k) if (sc[k] > mx_v) { mx_v = sc[k]; mx_i = k; }
    bool any_nz = false; int nz_i = 0; float nz_v = -3.0e38f;
#pragma unroll
    for (int k = 0; k < K_; ++k)
      if (sc[k] != 0.f) { if (!any_nz || sc[k] > nz_v) { nz_v = sc[k]; nz_i = k; } any_nz = true; }

    bool hop      = (mx_v == 0.f) && !dead;
    int  sel_i    = (hop && any_nz) ? nz_i : mx_i;
    bool new_dead = dead || (hop && !any_nz);
    int  nbr      = nbrt[(size_t)idx * K_ + sel_i];
    int  nidx     = (hop && any_nz) ? nbr : idx;

    const float* sel = vg + sel_i * D_;
    float zf = new_dead ? 0.f : validf;
    _Float16* orow = Xh + ((size_t)b * LP + (size_t)n * SLOTS_ + step) * DP;
#pragma unroll
    for (int i = 0; i < 13; ++i) {
      int d = lane + i * 32;
      if (d < D_) orow[d] = (_Float16)(sel[d] * zf);
    }
    idx = nidx; dead = new_dead;
  }
}

// --------------------------- QKV GEMM --------------------------------------
// One wave per 16x16 output tile: Y[512,400] = X[512,416] * W[416,400].
// Q/K stored per-head [b][h][row][dh] (dh padded to 32, zeros); V transposed
// per-head [b][h][dh][row] so the AV GEMM's B-fragments load contiguously.
__global__ __launch_bounds__(256) void qkv_gemm_kernel(
    const _Float16* __restrict__ Xh, const _Float16* __restrict__ Wt,
    _Float16* __restrict__ Qh, _Float16* __restrict__ Kh, _Float16* __restrict__ Vt) {
  int wave = (blockIdx.x * blockDim.x + threadIdx.x) >> 5;
  int lane = threadIdx.x & 31;
  int nt = wave % 25; wave /= 25;   // 25 N-tiles (400 cols)
  int mt = wave % 32; wave /= 32;   // 32 M-tiles (512 rows)
  int w  = wave % 3;                // 0=Q 1=K 2=V
  int b  = wave / 3;

  const _Float16* A  = Xh + (size_t)b * LP * DP;
  const _Float16* Bt = Wt + (size_t)w * D_ * DP;

  v8f acc = {};
#pragma unroll
  for (int k0 = 0; k0 < DP; k0 += 32) {
    v16h a  = load_frag_A(A,  mt * 16, k0, DP);
    v16h bb = load_frag_B(Bt, k0, nt * 16, DP);
    if (k0 + 32 < DP) {
      __builtin_prefetch(A  + (size_t)(mt * 16 + (lane & 15)) * DP + k0 + 32, 0, 1);
      __builtin_prefetch(Bt + (size_t)(nt * 16 + (lane & 15)) * DP + k0 + 32, 0, 1);
    }
    acc = wmma_f16(a, bb, acc);
  }

  int ncol = nt * 16 + (lane & 15);           // 0..399
  int h = ncol / DH_, dh = ncol % DH_;
  int rbase = mt * 16 + ((lane >> 4) << 3);
  size_t hb = (size_t)b * H_ + h;
#pragma unroll
  for (int r = 0; r < 8; ++r) {
    int row = rbase + r;
    _Float16 v = (_Float16)acc[r];
    if (w == 0)      Qh[(hb * LP + row) * DHP + dh] = v;   // scale folded into Wq
    else if (w == 1) Kh[(hb * LP + row) * DHP + dh] = v;
    else             Vt[(hb * DHP + dh) * LP + row] = v;
  }
}

// --------------------------- attention -------------------------------------
// Block = (b, h, 16-query-row tile). S = Q*K^T via WMMA (Kdim=32 padded head),
// masked softmax in LDS, then O = P*V with K split over waves + LDS reduce.
__global__ __launch_bounds__(256) void attn_kernel(
    const _Float16* __restrict__ Qh, const _Float16* __restrict__ Kh,
    const _Float16* __restrict__ Vt, const int* __restrict__ mask,
    float* __restrict__ xattn) {
  __shared__ __align__(16) float    Sl[16][LP];       // 32 KB
  __shared__ __align__(16) _Float16 Pl[16][LP];       // 16 KB
  __shared__ __align__(16) float    Opart[8][16][16]; //  8 KB

  int bi = blockIdx.x;
  int mt = bi & 31;
  int h  = (bi >> 5) % H_;
  int b  = bi / (32 * H_);
  int tid = threadIdx.x, lane = tid & 31, wv = tid >> 5;

  size_t hb = (size_t)b * H_ + h;
  const _Float16* Qb = Qh + hb * LP * DHP;
  const _Float16* Kb = Kh + hb * LP * DHP;
  const _Float16* Vb = Vt + hb * DHP * LP;

  // Phase 1: scores. Each wave does 4 of the 32 N-tiles.
  v16h aq = load_frag_A(Qb, mt * 16, 0, DHP);
#pragma unroll
  for (int j = 0; j < 4; ++j) {
    int nt = wv * 4 + j;
    v16h bk = load_frag_B(Kb, 0, nt * 16, DHP);  // K^T fragment = rows of K
    v8f s = {};
    s = wmma_f16(aq, bk, s);
    int col = nt * 16 + (lane & 15);
    bool ok = (col < L_) && (mask[(size_t)b * L_ + col] > 0);
    int rb = (lane >> 4) << 3;
#pragma unroll
    for (int r = 0; r < 8; ++r) Sl[rb + r][col] = ok ? s[r] : -1e9f;
  }
  __syncthreads();

  // Phase 2: softmax over 512 cols, 2 rows per wave; write P as f16.
#pragma unroll
  for (int rr = 0; rr < 2; ++rr) {
    int row = wv * 2 + rr;
    float mx = -3.0e38f;
#pragma unroll
    for (int j = 0; j < 16; ++j) mx = fmaxf(mx, Sl[row][lane + j * 32]);
#pragma unroll
    for (int m = 16; m >= 1; m >>= 1) mx = fmaxf(mx, __shfl_xor(mx, m, 32));
    float sum = 0.f;
#pragma unroll
    for (int j = 0; j < 16; ++j) sum += __expf(Sl[row][lane + j * 32] - mx);
    sum = waveSum(sum);
    float inv = 1.f / sum;
#pragma unroll
    for (int j = 0; j < 16; ++j) {
      int c = lane + j * 32;
      Pl[row][c] = (_Float16)(__expf(Sl[row][c] - mx) * inv);
    }
  }
  __syncthreads();

  // Phase 3: O = P x V. wave -> (dim-tile nt2 = wv&1, K quarter kq = wv>>1).
  int nt2 = wv & 1, kq = wv >> 1;
  v8f acc = {};
#pragma unroll
  for (int s = 0; s < 4; ++s) {
    int k0 = (kq * 4 + s) * 32;
    v16h ap = load_frag_A(&Pl[0][0], 0, k0, LP);        // ds_load fragments
    v16h bv = load_frag_B(Vb, k0, nt2 * 16, LP);        // V^T fragments
    acc = wmma_f16(ap, bv, acc);
  }
  {
    int c = lane & 15, rb = (lane >> 4) << 3;
#pragma unroll
    for (int r = 0; r < 8; ++r) Opart[wv][rb + r][c] = acc[r];
  }
  __syncthreads();

  // Phase 4: reduce K-quarters and scatter valid head dims (dh < 20).
  int row = tid >> 4, c = tid & 15;
#pragma unroll
  for (int t2 = 0; t2 < 2; ++t2) {
    float v = Opart[t2][row][c] + Opart[t2 + 2][row][c] +
              Opart[t2 + 4][row][c] + Opart[t2 + 6][row][c];
    int dim = t2 * 16 + c;
    if (dim < DH_)
      xattn[((size_t)b * LP + mt * 16 + row) * D_ + h * DH_ + dim] = v;
  }
}

// ------------------- LN1 + attention pool + LN2 -----------------------------
__global__ __launch_bounds__(256) void ln_pool_kernel(
    const float* __restrict__ xattn,
    const float* __restrict__ ln1g, const float* __restrict__ ln1b,
    const float* __restrict__ fc1w, const float* __restrict__ fc1b,
    const float* __restrict__ fc2w, const float* __restrict__ fc2b,
    const float* __restrict__ ln2g, const float* __restrict__ ln2b,
    const int* __restrict__ mask, float* __restrict__ out) {
  __shared__ float xln[SLOTS_][D_];   // 16 KB
  __shared__ float pool[D_];
  __shared__ float sredA[8], sredB[8];
  __shared__ float sstat[2];
  __shared__ float ev[SLOTS_], av[SLOTS_];

  int bn = blockIdx.x;                 // (b * 50 + n)
  int b = bn / NN_, n = bn % NN_;
  int tid = threadIdx.x, lane = tid & 31, wv = tid >> 5;

  // LN1 per slot row
  for (int s = 0; s < SLOTS_; ++s) {
    const float* xr = xattn + ((size_t)b * LP + n * SLOTS_ + s) * D_;
    float p = 0.f, q = 0.f;
    for (int d = tid; d < D_; d += 256) { float v = xr[d]; p += v; q += v * v; }
    p = waveSum(p); q = waveSum(q);
    if (lane == 0) { sredA[wv] = p; sredB[wv] = q; }
    __syncthreads();
    if (tid == 0) {
      float sp = 0.f, sq = 0.f;
      for (int i = 0; i < 8; ++i) { sp += sredA[i]; sq += sredB[i]; }
      float mean = sp / D_;
      sstat[0] = mean;
      sstat[1] = rsqrtf(sq / D_ - mean * mean + 1e-5f);
    }
    __syncthreads();
    float mean = sstat[0], inv = sstat[1];
    for (int d = tid; d < D_; d += 256)
      xln[s][d] = (xr[d] - mean) * inv * ln1g[d] + ln1b[d];
    __syncthreads();
  }

  // e_s = tanh(xln @ fc1 + b1) @ fc2 + b2   (threads 0..127 own hidden units)
  for (int s = 0; s < SLOTS_; ++s) {
    float contrib = 0.f;
    if (tid < ATTH) {
      float a = fc1b[tid];
      for (int d = 0; d < D_; ++d) a += xln[s][d] * fc1w[(size_t)d * ATTH + tid];
      contrib = tanhf(a) * fc2w[tid];
    }
    contrib = waveSum(contrib);
    if (lane == 0) sredA[wv] = contrib;
    __syncthreads();
    if (tid == 0) {
      float e = fc2b[0];
      for (int i = 0; i < 8; ++i) e += sredA[i];
      int mk = mask[(size_t)b * L_ + n * SLOTS_ + s];
      ev[s] = (mk > 0) ? e : -1e9f;
    }
    __syncthreads();
  }

  if (tid == 0) {
    float mx = ev[0];
    for (int s = 1; s < SLOTS_; ++s) mx = fmaxf(mx, ev[s]);
    float sum = 0.f;
    for (int s = 0; s < SLOTS_; ++s) { av[s] = __expf(ev[s] - mx); sum += av[s]; }
    float inv = 1.f / sum;
    for (int s = 0; s < SLOTS_; ++s) av[s] *= inv;
  }
  __syncthreads();

  // pooled vector + LN2
  float p = 0.f, q = 0.f;
  for (int d = tid; d < D_; d += 256) {
    float v = 0.f;
#pragma unroll
    for (int s = 0; s < SLOTS_; ++s) v += av[s] * xln[s][d];
    pool[d] = v; p += v; q += v * v;
  }
  p = waveSum(p); q = waveSum(q);
  if (lane == 0) { sredA[wv] = p; sredB[wv] = q; }
  __syncthreads();
  if (tid == 0) {
    float sp = 0.f, sq = 0.f;
    for (int i = 0; i < 8; ++i) { sp += sredA[i]; sq += sredB[i]; }
    float mean = sp / D_;
    sstat[0] = mean;
    sstat[1] = rsqrtf(sq / D_ - mean * mean + 1e-5f);
  }
  __syncthreads();
  float mean = sstat[0], inv = sstat[1];
  for (int d = tid; d < D_; d += 256)
    out[(size_t)bn * D_ + d] = (pool[d] - mean) * inv * ln2g[d] + ln2b[d];
}

// ------------------------------ launcher -----------------------------------

extern "C" void kernel_launch(void* const* d_in, const int* in_sizes, int n_in,
                              void* d_out, int out_size, void* d_ws, size_t ws_size,
                              hipStream_t stream) {
  (void)in_sizes; (void)n_in; (void)out_size; (void)ws_size;
  const float* news  = (const float*)d_in[0];
  const float* table = (const float*)d_in[1];
  const float* Wq    = (const float*)d_in[2];
  const float* Wk    = (const float*)d_in[3];
  const float* Wv    = (const float*)d_in[4];
  const float* ln1g  = (const float*)d_in[5];
  const float* ln1b  = (const float*)d_in[6];
  const float* fc1w  = (const float*)d_in[7];
  const float* fc1b  = (const float*)d_in[8];
  const float* fc2w  = (const float*)d_in[9];
  const float* fc2b  = (const float*)d_in[10];
  const float* ln2g  = (const float*)d_in[11];
  const float* ln2b  = (const float*)d_in[12];
  const int*   click = (const int*)d_in[13];
  const int*   nbrt  = (const int*)d_in[14];
  const int*   mask  = (const int*)d_in[15];
  float*       out   = (float*)d_out;

  // Workspace layout (halves): Xh | Qh | Kh | Vt | Wt | xattn(f32). ~52 MB.
  _Float16* base = (_Float16*)d_ws;
  const size_t nX = (size_t)B_ * LP * DP;            // 3,407,872
  const size_t nQ = (size_t)B_ * H_ * LP * DHP;      // 5,242,880 each
  const size_t nW = 3ull * D_ * DP;                  //   499,200
  _Float16* Xh = base;
  _Float16* Qh = Xh + nX;
  _Float16* Kh = Qh + nQ;
  _Float16* Vt = Kh + nQ;
  _Float16* Wt = Vt + nQ;
  float*    xattn = (float*)(Wt + nW);               // 16*512*400 f32

  // 1) zero f16 buffers (covers slot/row/col/head-dim padding)
  zero_h_kernel<<<1024, 256, 0, stream>>>(Xh, (long long)(nX + 3 * nQ));
  // 2) transpose+pad weights to f16 (fold 1/sqrt(DH) into Wq)
  convw_kernel<<<(int)((nW + 255) / 256), 256, 0, stream>>>(Wq, Wk, Wv, Wt);
  // 3) graph traversal: 800 waves (one per (b,n))
  traverse_kernel<<<(B_ * NN_ * 32 + 255) / 256, 256, 0, stream>>>(news, table, click, nbrt, Xh);
  // 4) QKV projection: 16 batches * 3 mats * 32 * 25 tiles = 38400 waves
  qkv_gemm_kernel<<<(B_ * 3 * 32 * 25) / 8, 256, 0, stream>>>(Xh, Wt, Qh, Kh, Vt);
  // 5) attention: 16 * 20 heads * 32 query tiles
  attn_kernel<<<B_ * H_ * (LP / 16), 256, 0, stream>>>(Qh, Kh, Vt, mask, xattn);
  // 6) LN1 + attention pooling + LN2 -> d_out
  ln_pool_kernel<<<B_ * NN_, 256, 0, stream>>>(xattn, ln1g, ln1b, fc1w, fc1b,
                                               fc2w, fc2b, ln2g, ln2b, mask, out);
}
